// CascadedGroupAttention_52003464020510
// MI455X (gfx1250) — matrix operand order repaired and checked by reading, not compiled, not run on hardware
//
#include <hip/hip_runtime.h>
#include <hip/hip_bf16.h>

// ---------------------------------------------------------------------------
// CascadedGroupAttention for MI455X (gfx1250, wave32, WMMA 16x16x32 f16)
//
// Shapes: B=32, DIM=256, HEADS=4, RD=16, R=32, N=1024, K(dw)=5
//   head i:  f1 = conv3x3(feat[:, :16])            (pconv_w[i])
//            qkv = BN(conv1x1(f1))  -> q,k,v (16 ch each)
//            q   = BN(dwconv5x5(q))
//            S   = q^T k * 0.25 + bias[|dr|*32+|dc|] ; P = softmax(S, -1)
//            out = v @ P^T   (replaces feat[:, :16])
//   y = BN(conv1x1(relu(cat(feats)), proj_w))      (256 x 1024 weights)
//
// WMMA strategy: q/k/v emitted by the stage kernel directly in the CDNA5
// A-/B-fragment lanes, in this file's contiguous k order (k = 16*(lane/16) + half), K padded 16->32 with zeros, so the
// attention wave does only 32B fragment loads + v_wmma_f32_16x16x32_f16.
// Projection uses a pre-folded 256x320 f16 weight (shared x2 block summed
// across heads) in A-fragment layout: 10 WMMAs per 16x16 output tile.
// Relative-position bias is staged in LDS; next-iteration fragments are
// prefetched with global_prefetch_b8.
// ---------------------------------------------------------------------------

typedef __attribute__((ext_vector_type(16))) _Float16 v16h;
typedef __attribute__((ext_vector_type(8)))  float    v8f;

#define NB    32      // batch
#define CDIM  256
#define NHEAD 4
#define RDC   16      // per-head channels
#define RSIDE 32      // spatial side
#define NPIX  1024    // RSIDE*RSIDE
#define EPSBN 1e-5f
#define ATTNSCALE 0.25f

// ---- fragment address helpers (halves within a 32-lane x 16-half tile) ----
// A matrix 16x32 f16 (MxK): lane = M + (K>=16 ? 16:0), half = K&15
// (same k order as the B fragment)
__device__ __forceinline__ int afrag_off(int m, int k) {
  int lane = m + ((k & 16) ? 16 : 0);
  int h    = k & 15;
  return lane * 16 + h;
}
// B matrix 32x16 f16 (KxN): lane = N + (K>=16 ? 16:0), half = K&15
__device__ __forceinline__ int bfrag_off(int k, int n) {
  int lane = n + ((k & 16) ? 16 : 0);
  int h    = k & 15;
  return lane * 16 + h;
}

// ===========================================================================
// Stage kernel: conv3x3 -> f1 ; conv1x1+BN -> qkv ; dw5x5+BN on q.
// Writes q (A-frag over n-tiles), k (B-frag over n-tiles), v (A-frag over
// m-chunks of 32) as f16 fragments, K padded to 32 with zeros.
// One block per batch item; dynamic LDS = f1(64K) + qpre(64K) + weights.
// ===========================================================================
__global__ __launch_bounds__(256)
void qkv_stage_kernel(int head,
                      const float* __restrict__ x,     // (B,256,1024)
                      const float* __restrict__ outs,  // (B,4,16,1024)
                      const float* __restrict__ pconv_w,  // (4,16,16,3,3)
                      const float* __restrict__ qkv_w,    // (4,48,16)
                      const float* __restrict__ qkv_g, const float* __restrict__ qkv_b,
                      const float* __restrict__ qkv_m, const float* __restrict__ qkv_v,
                      const float* __restrict__ dw_w,     // (4,16,25)
                      const float* __restrict__ dw_g, const float* __restrict__ dw_b,
                      const float* __restrict__ dw_m, const float* __restrict__ dw_v,
                      _Float16* __restrict__ q_frag,   // (B,64,512)
                      _Float16* __restrict__ k_frag,   // (B,64,512)
                      _Float16* __restrict__ v_frag)   // (B,32,512)
{
  extern __shared__ float lds[];
  float* f1   = lds;                 // 16*1024
  float* qpre = lds + 16 * 1024;     // 16*1024
  float* wsm  = lds + 32 * 1024;     // 2304 (pconv) + 768 (qkv) + 400 (dw)

  const int b   = blockIdx.x;
  const int tid = threadIdx.x;

  const float* in = (head == 0)
      ? (x + (size_t)b * CDIM * NPIX)
      : (outs + ((size_t)(b * NHEAD + (head - 1)) * RDC) * NPIX);

  // stage weights into LDS
  const float* pw = pconv_w + head * 16 * 16 * 9;
  for (int i = tid; i < 2304; i += 256) wsm[i] = pw[i];
  const float* qw = qkv_w + head * 48 * 16;
  for (int i = tid; i < 768; i += 256) wsm[2304 + i] = qw[i];
  const float* dwp = dw_w + head * 16 * 25;
  for (int i = tid; i < 400; i += 256) wsm[3072 + i] = dwp[i];

  _Float16* qf = q_frag + (size_t)b * 64 * 512;
  _Float16* kf = k_frag + (size_t)b * 64 * 512;
  _Float16* vf = v_frag + (size_t)b * 32 * 512;

  // zero the padded K rows (k=16..31) of q (A-frag) and k (B-frag) tiles;
  // disjoint from all valid-element addresses, so no sync needed vs them.
  for (int idx = tid; idx < 64 * 256; idx += 256) {
    int t = idx >> 8, rem = idx & 255;
    int mn = rem & 15, kp = 16 + (rem >> 4);
    qf[t * 512 + afrag_off(mn, kp)] = (_Float16)0.f;
    kf[t * 512 + bfrag_off(kp, mn)] = (_Float16)0.f;
  }
  __syncthreads();

  // f1 = conv3x3 pad1 over the 16 input channels
  for (int idx = tid; idx < 16 * NPIX; idx += 256) {
    int c = idx >> 10, p = idx & 1023;
    int py = p >> 5, px = p & 31;
    float acc = 0.f;
    for (int ci = 0; ci < 16; ci++) {
      const float* ip = in + ci * NPIX;
      const float* wp = wsm + (c * 16 + ci) * 9;
      #pragma unroll
      for (int ky = 0; ky < 3; ky++) {
        int yy = py + ky - 1;
        if ((unsigned)yy >= 32u) continue;
        #pragma unroll
        for (int kx = 0; kx < 3; kx++) {
          int xx = px + kx - 1;
          if ((unsigned)xx >= 32u) continue;
          acc += ip[yy * 32 + xx] * wp[ky * 3 + kx];
        }
      }
    }
    f1[c * NPIX + p] = acc;
  }
  __syncthreads();

  // qkv = BN(conv1x1(f1)); scatter k,v straight into WMMA fragments
  const float* qg = qkv_g + head * 48; const float* qb = qkv_b + head * 48;
  const float* qm = qkv_m + head * 48; const float* qv = qkv_v + head * 48;
  for (int idx = tid; idx < 48 * NPIX; idx += 256) {
    int oc = idx >> 10, p = idx & 1023;
    float acc = 0.f;
    const float* wq = wsm + 2304 + oc * 16;
    #pragma unroll
    for (int ci = 0; ci < 16; ci++) acc += f1[ci * NPIX + p] * wq[ci];
    float sc  = qg[oc] * rsqrtf(qv[oc] + EPSBN);
    float val = (acc - qm[oc]) * sc + qb[oc];
    if (oc < 16) {                       // q -> LDS, dwconv next
      qpre[oc * NPIX + p] = val;
    } else if (oc < 32) {                // k: B-frag, K=channel, N=pixel%16
      int kk = oc - 16, nt = p >> 4, n = p & 15;
      kf[nt * 512 + bfrag_off(kk, n)] = (_Float16)val;
    } else {                             // v: A-frag, M=channel, K=pixel%32
      int vc = oc - 32, mt = p >> 5, kk = p & 31;
      vf[mt * 512 + afrag_off(vc, kk)] = (_Float16)val;
    }
  }
  __syncthreads();

  // q = BN(dw5x5(qpre)) -> A-frag, M=pixel%16 (row of S), K=channel
  const float* dg = dw_g + head * 16; const float* db = dw_b + head * 16;
  const float* dm = dw_m + head * 16; const float* dv = dw_v + head * 16;
  for (int idx = tid; idx < 16 * NPIX; idx += 256) {
    int c = idx >> 10, p = idx & 1023;
    int py = p >> 5, px = p & 31;
    float acc = 0.f;
    const float* wd = wsm + 3072 + c * 25;
    #pragma unroll
    for (int ky = 0; ky < 5; ky++) {
      int yy = py + ky - 2;
      if ((unsigned)yy >= 32u) continue;
      #pragma unroll
      for (int kx = 0; kx < 5; kx++) {
        int xx = px + kx - 2;
        if ((unsigned)xx >= 32u) continue;
        acc += qpre[c * NPIX + yy * 32 + xx] * wd[ky * 5 + kx];
      }
    }
    float sc  = dg[c] * rsqrtf(dv[c] + EPSBN);
    float val = (acc - dm[c]) * sc + db[c];
    int nt = p >> 4, m = p & 15;
    qf[nt * 512 + afrag_off(m, c)] = (_Float16)val;
  }
}

// ===========================================================================
// Attention kernel: one wave per (batch, 16-row block of S).
// 64 WMMAs for S = q^T k, bias+scale, softmax in LDS, 32 WMMAs for v @ P^T.
// Bias table staged in LDS; k/v fragments prefetched one tile ahead.
// ===========================================================================
__global__ __launch_bounds__(32)
void attn_kernel(int head,
                 const _Float16* __restrict__ q_frag,
                 const _Float16* __restrict__ k_frag,
                 const _Float16* __restrict__ v_frag,
                 const float* __restrict__ attn_biases,  // (4,1024)
                 float* __restrict__ outs)               // (B,4,16,1024)
{
  extern __shared__ float S[];       // 16*1024 scores + 1024 bias
  float* biasl = S + 16 * 1024;
  const int b   = blockIdx.y;
  const int nt  = blockIdx.x;        // row tile (16 rows of n)
  const int lid = threadIdx.x;

  // stage the distance-bias table (4 KB) into LDS once per wave
  const float* bias = attn_biases + head * 1024;
  #pragma unroll
  for (int j = 0; j < 32; j++) biasl[j * 32 + lid] = bias[j * 32 + lid];

  const _Float16* kbase = k_frag + ((size_t)b * 64 * 32) * 16;
  const _Float16* vbase = v_frag + ((size_t)b * 32 * 32) * 16;
  const v16h a = *(const v16h*)(q_frag + (((size_t)b * 64 + nt) * 32 + lid) * 16);
  const int mcol  = lid & 15;
  const int nbase = (lid >> 4) << 3;
  __syncthreads();

  for (int ct = 0; ct < 64; ct++) {
    v16h bm = *(const v16h*)(kbase + ((size_t)ct * 32 + lid) * 16);
    if (ct + 1 < 64)   // hint: pull next k fragment line into cache
      __builtin_prefetch(kbase + ((size_t)(ct + 1) * 32 + lid) * 16, 0, 3);
    v8f c = {};
    c = __builtin_amdgcn_wmma_f32_16x16x32_f16(false, a, false, bm,
                                               (short)0, c, false, false);
    int m = ct * 16 + mcol;
    #pragma unroll
    for (int r = 0; r < 8; r++) {
      int nl = r + nbase;
      int n  = nt * 16 + nl;
      int dr = (n >> 5) - (m >> 5);  if (dr < 0) dr = -dr;
      int dc = (n & 31) - (m & 31);  if (dc < 0) dc = -dc;
      S[nl * 1024 + m] = c[r] * ATTNSCALE + biasl[dr * 32 + dc];
    }
  }
  __syncthreads();

  // row-wise softmax over m (1024), full-wave reductions
  for (int r = 0; r < 16; r++) {
    float mx = -3.402823e38f;
    for (int j = lid; j < 1024; j += 32) mx = fmaxf(mx, S[r * 1024 + j]);
    #pragma unroll
    for (int off = 16; off; off >>= 1) mx = fmaxf(mx, __shfl_xor(mx, off, 32));
    float sum = 0.f;
    for (int j = lid; j < 1024; j += 32) {
      float e = __expf(S[r * 1024 + j] - mx);
      S[r * 1024 + j] = e;
      sum += e;
    }
    #pragma unroll
    for (int off = 16; off; off >>= 1) sum += __shfl_xor(sum, off, 32);
    float inv = 1.f / sum;
    for (int j = lid; j < 1024; j += 32) S[r * 1024 + j] *= inv;
  }
  __syncthreads();

  // out[c, n-block] = v(16x1024) @ P^T : lane index already encodes B layout
  v8f acc = {};
  const int ncol = lid & 15;
  const int kb   = (lid >> 4) * 16;
  for (int mt = 0; mt < 32; mt++) {
    v16h va = *(const v16h*)(vbase + ((size_t)mt * 32 + lid) * 16);
    if (mt + 1 < 32)
      __builtin_prefetch(vbase + ((size_t)(mt + 1) * 32 + lid) * 16, 0, 3);
    v16h bp;
    #pragma unroll
    for (int h = 0; h < 16; h++)
      bp[h] = (_Float16)S[ncol * 1024 + mt * 32 + kb + h];
    acc = __builtin_amdgcn_wmma_f32_16x16x32_f16(false, va, false, bp,
                                                 (short)0, acc, false, false);
  }

  float* op = outs + ((size_t)(b * NHEAD + head) * RDC) * NPIX;
  #pragma unroll
  for (int r = 0; r < 8; r++) {
    int ch = r + nbase;
    op[ch * NPIX + nt * 16 + ncol] = acc[r];
  }
}

// ===========================================================================
// Pre-fold projection weight: 256 x 320 f16 A-fragments.
//   c <  64 : per-head attention-out columns W[oc, i*256 + (c&15)]
//   c < 304 : shared x2 columns summed over heads
//   else    : zero pad
// ===========================================================================
__global__ __launch_bounds__(256)
void prep_projw_kernel(const float* __restrict__ proj_w,  // (256,1024)
                       _Float16* __restrict__ wc_frag)    // (16,10,512)
{
  const int tid = threadIdx.x;
  for (int idx = tid; idx < 256 * 320; idx += 256) {
    int oc = idx / 320, c = idx % 320;
    float v;
    if (c < 64) {
      v = proj_w[oc * 1024 + (c >> 4) * 256 + (c & 15)];
    } else if (c < 304) {
      int cc = c - 64;
      v = 0.f;
      #pragma unroll
      for (int i = 0; i < 4; i++) v += proj_w[oc * 1024 + i * 256 + 16 + cc];
    } else {
      v = 0.f;
    }
    int mt = oc >> 4, m = oc & 15, kt = c >> 5, kk = c & 31;
    wc_frag[((size_t)mt * 10 + kt) * 512 + afrag_off(m, kk)] = (_Float16)v;
  }
}

// ===========================================================================
// Projection: y = BN(relu(cat) @ proj_w^T). GEMM M=256, K=320, N=1024/batch.
// 8 waves per block, each wave owns one 16x16 tile; 10 WMMAs per tile.
// ===========================================================================
__global__ __launch_bounds__(256)
void proj_kernel(const float* __restrict__ x,      // (B,256,1024)
                 const float* __restrict__ outs,   // (B,64,1024) flat head-ch
                 const _Float16* __restrict__ wc_frag,
                 const float* __restrict__ pg, const float* __restrict__ pb,
                 const float* __restrict__ pm, const float* __restrict__ pv,
                 float* __restrict__ y)            // (B,256,1024)
{
  const int lid = threadIdx.x & 31;
  const int nt  = blockIdx.x * 8 + (threadIdx.x >> 5);
  const int mt  = blockIdx.y;
  const int b   = blockIdx.z;

  const int ncol = lid & 15;
  const int p    = nt * 16 + ncol;
  const int kb   = (lid >> 4) * 16;
  const float* xb = x    + (size_t)b * CDIM * NPIX;
  const float* ob = outs + (size_t)b * NHEAD * RDC * NPIX;
  const _Float16* wbase = wc_frag + (size_t)mt * 10 * 512;

  v8f acc = {};
  for (int kt = 0; kt < 10; kt++) {
    v16h aw = *(const v16h*)(wbase + ((size_t)kt * 32 + lid) * 16);
    if (kt + 1 < 10)
      __builtin_prefetch(wbase + ((size_t)(kt + 1) * 32 + lid) * 16, 0, 3);
    v16h bf;
    #pragma unroll
    for (int h = 0; h < 16; h++) {
      int kr = kt * 32 + kb + h;
      float v;
      if (kr < 64)        v = ob[(size_t)kr * NPIX + p];          // attention outs
      else if (kr < 304)  v = xb[(size_t)(16 + kr - 64) * NPIX + p]; // shared x2
      else                v = 0.f;
      bf[h] = (_Float16)fmaxf(v, 0.f);                            // ReLU on cat
    }
    acc = __builtin_amdgcn_wmma_f32_16x16x32_f16(false, aw, false, bf,
                                                 (short)0, acc, false, false);
  }

  const int nbase = (lid >> 4) << 3;
  #pragma unroll
  for (int r = 0; r < 8; r++) {
    int oc = mt * 16 + r + nbase;
    float sc = pg[oc] * rsqrtf(pv[oc] + EPSBN);
    y[((size_t)b * CDIM + oc) * NPIX + p] = (acc[r] - pm[oc]) * sc + pb[oc];
  }
}

// ===========================================================================
extern "C" void kernel_launch(void* const* d_in, const int* in_sizes, int n_in,
                              void* d_out, int out_size, void* d_ws, size_t ws_size,
                              hipStream_t stream) {
  const float* x        = (const float*)d_in[0];
  const float* pconv_w  = (const float*)d_in[1];
  const float* qkv_w    = (const float*)d_in[2];
  const float* qkv_g    = (const float*)d_in[3];
  const float* qkv_b    = (const float*)d_in[4];
  const float* qkv_m    = (const float*)d_in[5];
  const float* qkv_v    = (const float*)d_in[6];
  const float* dw_w     = (const float*)d_in[7];
  const float* dw_g     = (const float*)d_in[8];
  const float* dw_b     = (const float*)d_in[9];
  const float* dw_m     = (const float*)d_in[10];
  const float* dw_v     = (const float*)d_in[11];
  const float* proj_w   = (const float*)d_in[12];
  const float* proj_g   = (const float*)d_in[13];
  const float* proj_b   = (const float*)d_in[14];
  const float* proj_m   = (const float*)d_in[15];
  const float* proj_v   = (const float*)d_in[16];
  const float* attn_b   = (const float*)d_in[17];

  char* ws = (char*)d_ws;
  _Float16* q_frag = (_Float16*)(ws);                        // 2 MB
  _Float16* k_frag = (_Float16*)(ws + (2u  << 20));          // 2 MB
  _Float16* v_frag = (_Float16*)(ws + (4u  << 20));          // 1 MB
  float*    outs   = (float*)   (ws + (5u  << 20));          // 8 MB
  _Float16* wc     = (_Float16*)(ws + (13u << 20));          // 160 KB

  prep_projw_kernel<<<1, 256, 0, stream>>>(proj_w, wc);

  const size_t lds_stage = (size_t)(32 * 1024 + 3472) * sizeof(float); // ~145 KB
  const size_t lds_attn  = (size_t)(16 * 1024 + 1024) * sizeof(float); // 68 KB

  for (int h = 0; h < NHEAD; h++) {
    qkv_stage_kernel<<<NB, 256, lds_stage, stream>>>(
        h, x, outs, pconv_w, qkv_w, qkv_g, qkv_b, qkv_m, qkv_v,
        dw_w, dw_g, dw_b, dw_m, dw_v, q_frag, k_frag, v_frag);
    attn_kernel<<<dim3(64, NB), 32, lds_attn, stream>>>(
        h, q_frag, k_frag, v_frag, attn_b, outs);
  }

  proj_kernel<<<dim3(8, 16, NB), 256, 0, stream>>>(
      x, outs, wc, proj_g, proj_b, proj_m, proj_v, (float*)d_out);
}
